// ActorNetwork_20418274525777
// MI455X (gfx1250) — matrix operand order repaired and checked by reading, not compiled, not run on hardware
//
#include <hip/hip_runtime.h>

// MI455X / gfx1250 implementation of a 5-step Euler neural-ODE MLP.
// Roofline: ~2.8 TFLOP dominated by [B,2048]x[2048,2048] GEMM -> matrix cores.
// Strategy: f16 WMMA (f32 accumulate), batch chunked to 8192 rows so the
// h1/h2 activations (32MB f16 each) + f16 weights (~8.5MB) stay L2-resident.
// Hot GEMM uses a 32(M)x128(N) per-wave tile: 16 WMMAs per 12 vector loads.

typedef __attribute__((ext_vector_type(16))) _Float16 v16h;
typedef __attribute__((ext_vector_type(8)))  _Float16 v8h;
typedef __attribute__((ext_vector_type(8)))  float    v8f;

#define BATCH  65536
#define HID    2048
#define CHUNK  8192
#define NSTEPS 5
#define DT_F   0.2f

__device__ __forceinline__ v8f wmma16(v16h a, v16h b, v8f c) {
  // emits v_wmma_f32_16x16x32_f16
  return __builtin_amdgcn_wmma_f32_16x16x32_f16(
      /*neg_a=*/false, a, /*neg_b=*/false, b,
      /*c_mod=*/(short)0, c, /*reuse_a=*/false, /*reuse_b=*/false);
}

// A-fragment (16x32 f16). Lane holds row (lane&15). Halves 0..15 = K[hi*16..hi*16+15].
// p must already include +hi*16.
__device__ __forceinline__ v16h ldA(const _Float16* p) {
  union { v16h v; v8h h[2]; } u;
  u.h[0] = *(const v8h*)(p);
  u.h[1] = *(const v8h*)(p + 8);
  return u.v;
}

// B-fragment (32x16 f16) from pre-transposed W[N][K]. Lane holds column (lane&15);
// lo lanes K=0..15 contiguous, hi lanes K=16..31. p must include +hi*16.
__device__ __forceinline__ v16h ldB(const _Float16* p) {
  return *(const v16h*)p;
}

// A-fragment converted on the fly from an fp32 row-major source (K=32 padded state).
__device__ __forceinline__ v16h cvtA32(const float* p) {
  union { v16h v; v8h h[2]; } u;
#pragma unroll
  for (int i = 0; i < 8; ++i) {
    u.v[i]     = (_Float16)p[i];
    u.v[i + 8] = (_Float16)p[i + 8];
  }
  return u.v;
}

// ---------------- weight conversion (once per launch) ----------------

// W2 [K=2048][N=2048] fp32 -> W2t [N][K] f16, LDS-tiled transpose.
__global__ __launch_bounds__(256) void k_convW2t(const float* __restrict__ W2,
                                                 _Float16* __restrict__ W2t) {
  __shared__ float tile[32][33];
  const int tx = threadIdx.x & 31;
  const int ty = threadIdx.x >> 5;          // 0..7
  const int kBase = blockIdx.x * 32;
  const int nBase = blockIdx.y * 32;
#pragma unroll
  for (int i = 0; i < 32; i += 8)
    tile[ty + i][tx] = W2[(size_t)(kBase + ty + i) * HID + nBase + tx];
  __syncthreads();
#pragma unroll
  for (int i = 0; i < 32; i += 8)
    W2t[(size_t)(nBase + ty + i) * HID + kBase + tx] = (_Float16)tile[tx][ty + i];
}

// W1 [24][2048] fp32 -> W1t [2048][32] f16 (K padded 24->32 with zeros).
__global__ void k_convW1(const float* __restrict__ W1, _Float16* __restrict__ W1t) {
  int idx = blockIdx.x * 256 + threadIdx.x;          // 2048*32
  int n = idx >> 5, k = idx & 31;
  float v = (k < 24) ? W1[(size_t)k * HID + n] : 0.f;
  W1t[idx] = (_Float16)v;
}

// W3 [2048][24] fp32 -> W3t [32][2048] f16 (N padded 24->32 with zeros).
__global__ void k_convW3(const float* __restrict__ W3, _Float16* __restrict__ W3t) {
  int idx = blockIdx.x * 256 + threadIdx.x;          // 32*2048
  int n = idx >> 11, k = idx & 2047;
  float v = (n < 24) ? W3[(size_t)k * 24 + n] : 0.f;
  W3t[idx] = (_Float16)v;
}

// ---------------- state init + trajectory slot 0 ----------------

__global__ void k_init(const float* __restrict__ x, const float* __restrict__ z,
                       float* __restrict__ state, float* __restrict__ out) {
  int idx = blockIdx.x * 256 + threadIdx.x;          // BATCH*32
  int b = idx >> 5, c = idx & 31;
  float v = 0.f;
  if (c < 15)      v = x[(size_t)b * 15 + c];
  else if (c < 24) v = z[(size_t)b * 9 + (c - 15)];
  state[idx] = v;                                    // fp32 state, K padded to 32
  if (c < 24) {
    float tv = v;
    if (c >= 15) tv = fminf(fmaxf(tv, -2.f), 2.f);   // clip force part of traj
    out[(size_t)b * 144 + c] = tv;                   // traj[b][0][c]
  }
}

// ---------------- layer 1: h1 = relu(state @ W1 + b1), K=32 single WMMA ----------------

__global__ __launch_bounds__(256) void k_layer1(
    const float* __restrict__ state, const _Float16* __restrict__ W1t,
    const float* __restrict__ b1, _Float16* __restrict__ h1) {
  const int lane = threadIdx.x & 31;
  const int wave = threadIdx.x >> 5;
  const int m  = lane & 15;
  const int hi = lane >> 4;
  const int rowBase = blockIdx.x * 64 + (wave >> 2) * 32;
  const int colBase = blockIdx.y * 256 + (wave & 3) * 64;

  const float* pa = state + (size_t)(rowBase + m) * 32 + hi * 16;
  v16h a0 = cvtA32(pa);
  v16h a1 = cvtA32(pa + 16 * 32);
  const _Float16* pb = W1t + (size_t)(colBase + m) * 32 + hi * 16;

  v8f acc[2][4] = {};
#pragma unroll
  for (int ni = 0; ni < 4; ++ni) {
    v16h b = ldB(pb + (size_t)ni * 16 * 32);
    acc[0][ni] = wmma16(a0, b, acc[0][ni]);
    acc[1][ni] = wmma16(a1, b, acc[1][ni]);
  }
#pragma unroll
  for (int ni = 0; ni < 4; ++ni) {
    float bias = b1[colBase + ni * 16 + m];
#pragma unroll
    for (int mi = 0; mi < 2; ++mi) {
      _Float16* pc = h1 + (size_t)(rowBase + mi * 16 + hi * 8) * HID
                        + colBase + ni * 16 + m;
#pragma unroll
      for (int r = 0; r < 8; ++r) {
        float v = acc[mi][ni][r] + bias;
        v = v > 0.f ? v : 0.f;
        pc[(size_t)r * HID] = (_Float16)v;
      }
    }
  }
}

// ---------------- layer 2: h2 = relu(h1 @ W2 + b2) — the hot GEMM ----------------
// Per wave: 32(M) x 128(N) tile, K-loop of 32. 16 WMMAs per K-step against
// 12 vector loads (2 A-frags, 8 B-frags), everything L2-resident.

__global__ __launch_bounds__(256) void k_layer2(
    const _Float16* __restrict__ h1, const _Float16* __restrict__ W2t,
    const float* __restrict__ b2, _Float16* __restrict__ h2) {
  const int lane = threadIdx.x & 31;
  const int wave = threadIdx.x >> 5;
  const int m  = lane & 15;
  const int hi = lane >> 4;
  const int rowBase = blockIdx.x * 64 + (wave >> 2) * 32;     // 2 waves in M
  const int colBase = blockIdx.y * 512 + (wave & 3) * 128;    // 4 waves in N

  const _Float16* pa0 = h1 + (size_t)(rowBase + m) * HID + hi * 16;
  const _Float16* pa1 = pa0 + (size_t)16 * HID;
  const _Float16* pb  = W2t + (size_t)(colBase + m) * HID + hi * 16;

  v8f acc[2][8] = {};
  for (int k = 0; k < HID; k += 32) {
    v16h a0 = ldA(pa0 + k);
    v16h a1 = ldA(pa1 + k);
#pragma unroll
    for (int ni = 0; ni < 8; ++ni) {
      v16h b = ldB(pb + (size_t)ni * 16 * HID + k);
      acc[0][ni] = wmma16(a0, b, acc[0][ni]);
      acc[1][ni] = wmma16(a1, b, acc[1][ni]);
    }
  }
#pragma unroll
  for (int ni = 0; ni < 8; ++ni) {
    float bias = b2[colBase + ni * 16 + m];
#pragma unroll
    for (int mi = 0; mi < 2; ++mi) {
      _Float16* pc = h2 + (size_t)(rowBase + mi * 16 + hi * 8) * HID
                        + colBase + ni * 16 + m;
#pragma unroll
      for (int r = 0; r < 8; ++r) {
        float v = acc[mi][ni][r] + bias;
        v = v > 0.f ? v : 0.f;
        pc[(size_t)r * HID] = (_Float16)v;
      }
    }
  }
}

// ---------------- layer 3 + Euler update + traj write (fused) ----------------
// out = h2 @ W3 + b3 (N=24 padded to 32); s += DT*out; traj slot = clip(s).

__global__ __launch_bounds__(256) void k_layer3(
    const _Float16* __restrict__ h2, const _Float16* __restrict__ W3t,
    const float* __restrict__ b3, float* __restrict__ state,
    float* __restrict__ traj) {
  const int lane = threadIdx.x & 31;
  const int wave = threadIdx.x >> 5;
  const int m  = lane & 15;
  const int hi = lane >> 4;
  const int rowBase = blockIdx.x * 256 + wave * 32;

  const _Float16* pa0 = h2 + (size_t)(rowBase + m) * HID + hi * 16;
  const _Float16* pa1 = pa0 + (size_t)16 * HID;
  const _Float16* pb0 = W3t + (size_t)m * HID + hi * 16;
  const _Float16* pb1 = W3t + (size_t)(16 + m) * HID + hi * 16;

  v8f acc[2][2] = {};
  for (int k = 0; k < HID; k += 32) {
    v16h a0 = ldA(pa0 + k);
    v16h a1 = ldA(pa1 + k);
    v16h b0 = ldB(pb0 + k);
    v16h b1 = ldB(pb1 + k);
    acc[0][0] = wmma16(a0, b0, acc[0][0]);
    acc[1][0] = wmma16(a1, b0, acc[1][0]);
    acc[0][1] = wmma16(a0, b1, acc[0][1]);
    acc[1][1] = wmma16(a1, b1, acc[1][1]);
  }
#pragma unroll
  for (int ni = 0; ni < 2; ++ni) {
    int col = ni * 16 + m;
    if (col < 24) {
      float bias = b3[col];
#pragma unroll
      for (int mi = 0; mi < 2; ++mi) {
        int rb = rowBase + mi * 16 + hi * 8;
#pragma unroll
        for (int r = 0; r < 8; ++r) {
          int row = rb + r;
          float o  = acc[mi][ni][r] + bias;
          float sv = state[(size_t)row * 32 + col] + DT_F * o;
          state[(size_t)row * 32 + col] = sv;
          float tv = (col >= 15) ? fminf(fmaxf(sv, -2.f), 2.f) : sv;
          traj[(size_t)row * 144 + col] = tv;
        }
      }
    }
  }
}

// ---------------- finalize: final_state + force from traj[:,5,:] ----------------

__global__ void k_final(float* __restrict__ out) {
  int idx = blockIdx.x * 256 + threadIdx.x;      // BATCH*24
  int b = idx / 24, c = idx % 24;
  float v = out[(size_t)b * 144 + 120 + c];      // traj[b][5][c] (already clipped)
  out[(size_t)BATCH * 144 + (size_t)b * 24 + c] = v;
  if (c >= 15)
    out[(size_t)BATCH * 144 + (size_t)BATCH * 24 + (size_t)b * 9 + (c - 15)] = v;
}

extern "C" void kernel_launch(void* const* d_in, const int* in_sizes, int n_in,
                              void* d_out, int out_size, void* d_ws, size_t ws_size,
                              hipStream_t stream) {
  (void)in_sizes; (void)n_in; (void)out_size; (void)ws_size;
  const float* x  = (const float*)d_in[0];
  const float* z  = (const float*)d_in[1];
  const float* W1 = (const float*)d_in[2];
  const float* b1 = (const float*)d_in[3];
  const float* W2 = (const float*)d_in[4];
  const float* b2 = (const float*)d_in[5];
  const float* W3 = (const float*)d_in[6];
  const float* b3 = (const float*)d_in[7];
  float* out = (float*)d_out;

  // Workspace layout (~84.2 MB total), all 256B-aligned offsets:
  char* ws = (char*)d_ws;
  float*    state = (float*)(ws);                 //  8,388,608 B  (B x 32 fp32)
  _Float16* W2t   = (_Float16*)(ws + 8388608);    //  8,388,608 B
  _Float16* W1t   = (_Float16*)(ws + 16777216);   //    131,072 B
  _Float16* W3t   = (_Float16*)(ws + 16908288);   //    131,072 B
  _Float16* h1    = (_Float16*)(ws + 17039360);   // 33,554,432 B  (CHUNK x 2048 f16)
  _Float16* h2    = (_Float16*)(ws + 50593792);   // 33,554,432 B

  k_convW2t<<<dim3(HID / 32, HID / 32), 256, 0, stream>>>(W2, W2t);
  k_convW1 <<<(HID * 32) / 256, 256, 0, stream>>>(W1, W1t);
  k_convW3 <<<(32 * HID) / 256, 256, 0, stream>>>(W3, W3t);
  k_init   <<<(BATCH * 32) / 256, 256, 0, stream>>>(x, z, state, out);

  for (int c = 0; c < BATCH / CHUNK; ++c) {
    float* st    = state + (size_t)c * CHUNK * 32;
    float* trajc = out   + (size_t)c * CHUNK * 144;
    for (int s = 0; s < NSTEPS; ++s) {
      k_layer1<<<dim3(CHUNK / 64, HID / 256), 256, 0, stream>>>(st, W1t, b1, h1);
      k_layer2<<<dim3(CHUNK / 64, HID / 512), 256, 0, stream>>>(h1, W2t, b2, h2);
      k_layer3<<<CHUNK / 256, 256, 0, stream>>>(h2, W3t, b3, st, trajc + (s + 1) * 24);
    }
  }
  k_final<<<(BATCH * 24) / 256, 256, 0, stream>>>(out);
}
